// CausalSelfAttention_77129022701872
// MI455X (gfx1250) — matrix operand order repaired and checked by reading, not compiled, not run on hardware
//
#include <hip/hip_runtime.h>
#include <math.h>

typedef __bf16 bf16;
typedef __attribute__((ext_vector_type(16))) __bf16 v16bf;
typedef __attribute__((ext_vector_type(8)))  __bf16 v8bf;
typedef __attribute__((ext_vector_type(8)))  float  v8f;
typedef __attribute__((ext_vector_type(4)))  unsigned u32x4;
typedef __attribute__((ext_vector_type(8)))  unsigned u32x8;

#define B_  2
#define S_  2048
#define E_  2048
#define H_  16
#define HD_ 128
#define SCALE_ 0.08838834764831845f  // 1/sqrt(128)

// LDS row strides (bf16 elements). Multiples of 8 (16B alignment for b128);
// 40 el = 80 B -> bank stride 20 (conflict-free over 16 lanes);
// 136 el = 272 B -> bank stride 68 = 4 mod 64 (conflict-free over 16 lanes).
#define LDA_S 40
#define LDB_S 40
#define LDK_S 136
#define LDP_S 40

// ---------------------------------------------------------------------------
// WMMA helpers (CDNA5 gfx1250, wave32). RA/RB = operand-reuse hints, legal
// when the A/B operand matches the previous (identical) WMMA instruction.
// ---------------------------------------------------------------------------
template <bool RA, bool RB>
__device__ inline v8f wmma_bf16(v16bf a, v16bf b, v8f c) {
  return __builtin_amdgcn_wmma_f32_16x16x32_bf16(
      /*neg_a=*/false, a, /*neg_b=*/false, b,
      /*c_mod=*/(short)0, c, /*reuse_a=*/RA, /*reuse_b=*/RB);
}

// A fragment: 16x32 (MxK), src row-major [r][k], leading dim ld.
// Per-lane data = 2 contiguous 8-element chunks -> 2x ds_load_b128.
__device__ inline v16bf load_a_frag(const bf16* src, int ld) {
  const int lane = threadIdx.x & 31;
  const int r = lane & 15;
  const int half = lane >> 4;
  v16bf a;
#pragma unroll
  for (int e = 0; e < 16; ++e) {
    const int vg = e >> 1, pos = e & 1;
    const int k = (half << 4) + (vg << 1) + pos;
    a[e] = src[r * ld + k];
  }
  return a;
}

// B fragment from transposed (column-major) storage: logical B[k][n] at
// src[n*ld + k]; per-lane k = half*16 + e -> 16 contiguous elements.
__device__ inline v16bf load_bT_frag(const bf16* src, int ld) {
  const int lane = threadIdx.x & 31;
  const int n = lane & 15;
  const int half = lane >> 4;
  v16bf b;
#pragma unroll
  for (int e = 0; e < 16; ++e) {
    const int k = (half << 4) + e;
    b[e] = src[n * ld + k];
  }
  return b;
}

// ---------------------------------------------------------------------------
// CDNA5 async & tensor data movers
// ---------------------------------------------------------------------------
__device__ inline void async_copy_b128(const void* gptr, const bf16* lptr) {
  const unsigned loff = (unsigned)(size_t)lptr;  // LDS flat addr low 32 bits
  asm volatile("global_load_async_to_lds_b128 %0, %1, off"
               :: "v"(loff), "v"(gptr) : "memory");
}
__device__ inline void wait_async_zero() {
  asm volatile("s_wait_asynccnt 0x0" ::: "memory");
}
__device__ inline void wait_tensor_zero() {
#if __has_builtin(__builtin_amdgcn_s_wait_tensorcnt)
  __builtin_amdgcn_s_wait_tensorcnt(0);
#else
  asm volatile("s_wait_tensorcnt 0x0" ::: "memory");
#endif
}

// TDM 2D tile load: 32 rows x 128 bf16 (row stride 128 el) from a 2048x128
// tensor into LDS, with hardware padding of 4 DWORDs (8 bf16) per 64-DWORD
// row -> LDS row stride = 136 el = LDK_S. D# per ISA ch.8 (groups 0..3).
__device__ inline void tdm_load_k_tile(const void* gaddr, unsigned lds_addr) {
  const unsigned long long ga = (unsigned long long)gaddr;
  u32x4 g0;
  g0[0] = 1u;                                   // count=1, user descriptor
  g0[1] = lds_addr;                             // lds_addr (bytes)
  g0[2] = (unsigned)ga;                         // global_addr[31:0]
  g0[3] = (unsigned)((ga >> 32) & 0x1FFFFFFu)   // global_addr[56:32]
          | (2u << 30);                         // type=2 ("image")
  u32x8 g1;
  g1[0] = (1u << 16)                            // data_size: 1 -> 2 bytes
        | (1u << 20)                            // pad_enable
        | (5u << 22)                            // pad_interval: 2^(5+1)=64 DW
        | (3u << 25);                           // pad_amount: 4 DWORDs
  g1[1] = (128u << 16);                         // tensor_dim0[15:0]=128
  g1[2] = (2048u << 16);                        // dim0 hi=0 | tensor_dim1 lo=2048
  g1[3] = (128u << 16);                         // dim1 hi=0 | tile_dim0=128
  g1[4] = 32u;                                  // tile_dim1=32, tile_dim2=0
  g1[5] = 128u;                                 // tensor_dim0_stride = 128
  g1[6] = 0u;
  g1[7] = 0u;
  u32x4 g2 = {0u, 0u, 0u, 0u};                  // dims 2+ unused
  u32x4 g3 = {0u, 0u, 0u, 0u};
  asm volatile("tensor_load_to_lds %0, %1, %2, %3"
               :: "s"(g0), "s"(g1), "s"(g2), "s"(g3) : "memory");
}

// ---------------------------------------------------------------------------
// f32 -> bf16 convert
// ---------------------------------------------------------------------------
__global__ __launch_bounds__(256) void cvt_f32_bf16(const float* __restrict__ src,
                                                    bf16* __restrict__ dst, int n) {
  int i = blockIdx.x * blockDim.x + threadIdx.x;
  const int stride = gridDim.x * blockDim.x;
  for (; i < n; i += stride) dst[i] = (bf16)src[i];
}

// ---------------------------------------------------------------------------
// Shared GEMM mainloop: 64x64 block tile, 4 waves (2x2), each wave 32x32,
// k-step 32. A async-staged row-major; B staged transposed ([n][k]).
// ---------------------------------------------------------------------------
__device__ inline void gemm_tiles(const bf16* __restrict__ A, int lda,
                                  const bf16* __restrict__ Bm, int ldb, int Kdim,
                                  int m0, int n0, bf16* As, bf16* Bst,
                                  v8f acc[2][2]) {
  const int t = threadIdx.x;
  const int wave = t >> 5;
  const int wrow = wave >> 1;  // 0..1
  const int wcol = wave & 1;   // 0..1

  for (int k0 = 0; k0 < Kdim; k0 += 32) {
    // A tile: 64x32 bf16, row-major, 2x async b128 per thread
    {
      const int r = t >> 2;         // 0..31
      const int c = (t & 3) << 3;   // 0,8,16,24
      async_copy_b128(&A[(size_t)(m0 + r) * lda + k0 + c], &As[r * LDA_S + c]);
      async_copy_b128(&A[(size_t)(m0 + r + 32) * lda + k0 + c],
                      &As[(r + 32) * LDA_S + c]);
      if (k0 + 32 < Kdim)
        __builtin_prefetch(&A[(size_t)(m0 + r) * lda + k0 + 32 + c], 0, 1);
    }
    // B tile: 32x64 bf16, stored transposed: Bst[n][k]
    {
      const int r = t >> 3;         // k row 0..15
      const int c = (t & 7) << 3;   // n col 0..56
      v8bf b0 = *(const v8bf*)&Bm[(size_t)(k0 + r) * ldb + n0 + c];
      v8bf b1 = *(const v8bf*)&Bm[(size_t)(k0 + r + 16) * ldb + n0 + c];
#pragma unroll
      for (int j = 0; j < 8; ++j) {
        Bst[(c + j) * LDB_S + r] = b0[j];
        Bst[(c + j) * LDB_S + r + 16] = b1[j];
      }
      if (k0 + 32 < Kdim)
        __builtin_prefetch(&Bm[(size_t)(k0 + 32 + r) * ldb + n0 + c], 0, 1);
    }
    wait_async_zero();
    __syncthreads();

    v16bf a0 = load_a_frag(As + (wrow * 32 + 0) * LDA_S, LDA_S);
    v16bf a1 = load_a_frag(As + (wrow * 32 + 16) * LDA_S, LDA_S);
    v16bf b0 = load_bT_frag(Bst + (wcol * 32 + 0) * LDB_S, LDB_S);
    v16bf b1 = load_bT_frag(Bst + (wcol * 32 + 16) * LDB_S, LDB_S);
    // Ordered so each hinted op repeats the previous op's A or B operand.
    acc[0][0] = wmma_bf16<false, false>(a0, b0, acc[0][0]);
    acc[0][1] = wmma_bf16<true,  false>(a0, b1, acc[0][1]);  // reuse A (a0)
    acc[1][1] = wmma_bf16<false, true >(a1, b1, acc[1][1]);  // reuse B (b1)
    acc[1][0] = wmma_bf16<true,  false>(a1, b0, acc[1][0]);  // reuse A (a1)
    __syncthreads();
  }
}

// ---------------------------------------------------------------------------
// QKV GEMM: [4096,2048] x [2048,6144] + b_qkv -> Q/K/V bf16 in [B,H,S,HD]
// ---------------------------------------------------------------------------
__global__ __launch_bounds__(128) void gemm_qkv_kernel(
    const bf16* __restrict__ A, const bf16* __restrict__ Bm,
    const float* __restrict__ bias,
    bf16* __restrict__ Qb, bf16* __restrict__ Kb, bf16* __restrict__ Vb) {
  __shared__ bf16 As[64 * LDA_S];
  __shared__ bf16 Bst[64 * LDB_S];
  const v8f zero = {0.f, 0.f, 0.f, 0.f, 0.f, 0.f, 0.f, 0.f};
  v8f acc[2][2] = {{zero, zero}, {zero, zero}};

  const int m0 = blockIdx.y * 64;
  const int n0 = blockIdx.x * 64;
  gemm_tiles(A, E_, Bm, 3 * E_, E_, m0, n0, As, Bst, acc);

  const int lane = threadIdx.x & 31;
  const int wave = threadIdx.x >> 5;
  const int wrow = wave >> 1, wcol = wave & 1;
  const int n = lane & 15, half = lane >> 4;
#pragma unroll
  for (int mi = 0; mi < 2; ++mi)
#pragma unroll
    for (int ni = 0; ni < 2; ++ni)
#pragma unroll
      for (int i = 0; i < 8; ++i) {
        const int row = m0 + wrow * 32 + mi * 16 + half * 8 + i;
        const int col = n0 + wcol * 32 + ni * 16 + n;
        const float v = acc[mi][ni][i] + bias[col];
        const int which = col >> 11;        // 0=q 1=k 2=v
        const int e = col & (E_ - 1);
        const int h = e >> 7, d = e & 127;
        const int bb = row >> 11, srow = row & (S_ - 1);
        bf16* dst = (which == 0) ? Qb : ((which == 1) ? Kb : Vb);
        dst[(((size_t)(bb * H_ + h)) * S_ + srow) * HD_ + d] = (bf16)v;
      }
}

// ---------------------------------------------------------------------------
// Flash attention: one (b,h) + 64 q-rows per block; wave w owns 16 q-rows.
// K tiles DMA'd by the Tensor Data Mover (with HW padding); V tiles staged
// transposed for contiguous b128 fragment loads.
// ---------------------------------------------------------------------------
__global__ __launch_bounds__(128) void attn_kernel(
    const bf16* __restrict__ Qm, const bf16* __restrict__ Km,
    const bf16* __restrict__ Vm, bf16* __restrict__ Om) {
  __shared__ bf16 Ks[32 * LDK_S];       // [kv][d], TDM pad -> 136 el rows
  __shared__ bf16 Vst[128 * LDB_S];     // [d][kv], transposed
  __shared__ bf16 Ps[4][16 * LDP_S];    // per-wave P tile, row-major

  const int t = threadIdx.x;
  const int lane = t & 31;
  const int wave = t >> 5;
  const int n = lane & 15;
  const int half = lane >> 4;
  const int bh = blockIdx.y;            // b*H + h
  const int q0 = blockIdx.x * 64;
  const int qr0 = q0 + wave * 16;

  // Preload this wave's Q tile (16x128) as 4 A-fragments
  const bf16* Qbase = Qm + ((size_t)bh * S_ + qr0) * HD_;
  v16bf qf[4];
#pragma unroll
  for (int tt = 0; tt < 4; ++tt) qf[tt] = load_a_frag(Qbase + tt * 32, HD_);

  const v8f zero = {0.f, 0.f, 0.f, 0.f, 0.f, 0.f, 0.f, 0.f};
  v8f o[8];
#pragma unroll
  for (int c2 = 0; c2 < 8; ++c2) o[c2] = zero;
  float mrun[8], lrun[8];
#pragma unroll
  for (int i = 0; i < 8; ++i) { mrun[i] = -3.0e38f; lrun[i] = 0.f; }

  const bf16* Kb = Km + (size_t)bh * S_ * HD_;
  const bf16* Vb = Vm + (size_t)bh * S_ * HD_;
  const unsigned ks_lds = (unsigned)(size_t)&Ks[0];

  for (int jb = 0; jb < q0 + 64; jb += 32) {
    // K tile: one TDM descriptor load (wave 0), HW-padded into Ks
    if (wave == 0) {
      tdm_load_k_tile(&Kb[(size_t)jb * HD_], ks_lds);
      wait_tensor_zero();
    }
    // V tile: 32x128, normal loads + transposed scatter into Vst[d][kv]
    {
      const int r = t >> 4;          // 0..7
      const int c = (t & 15) << 3;   // 0..120
#pragma unroll
      for (int rr = 0; rr < 32; rr += 8) {
        v8bf vv = *(const v8bf*)&Vb[(size_t)(jb + r + rr) * HD_ + c];
#pragma unroll
        for (int j = 0; j < 8; ++j) Vst[(c + j) * LDB_S + r + rr] = vv[j];
      }
    }
    __syncthreads();

    // S = Q * K^T : two 16x16 C-tiles over 32 kv cols, K-dim = HD (4 steps)
    v8f s0 = zero, s1 = zero;
#pragma unroll
    for (int tt = 0; tt < 4; ++tt) {
      s0 = wmma_bf16<false, false>(qf[tt], load_bT_frag(Ks + tt * 32, LDK_S), s0);
      s1 = wmma_bf16<true,  false>(qf[tt],  // reuse A (qf[tt])
                                   load_bT_frag(Ks + 16 * LDK_S + tt * 32, LDK_S), s1);
    }

    // Online softmax (row m = half*8+i, columns striped over lanes 0..15 per half)
#pragma unroll
    for (int i = 0; i < 8; ++i) {
      const int qr = qr0 + half * 8 + i;
      float v0 = (jb + n <= qr) ? s0[i] * SCALE_ : -3.0e38f;
      float v1 = (jb + 16 + n <= qr) ? s1[i] * SCALE_ : -3.0e38f;
      float mx = fmaxf(v0, v1);
#pragma unroll
      for (int msk = 1; msk < 16; msk <<= 1) mx = fmaxf(mx, __shfl_xor(mx, msk, 32));
      const float mn = fmaxf(mrun[i], mx);
      const float al = __expf(mrun[i] - mn);
      const float p0 = __expf(v0 - mn);
      const float p1 = __expf(v1 - mn);
      float rs = p0 + p1;
#pragma unroll
      for (int msk = 1; msk < 16; msk <<= 1) rs += __shfl_xor(rs, msk, 32);
      lrun[i] = lrun[i] * al + rs;
      mrun[i] = mn;
#pragma unroll
      for (int c2 = 0; c2 < 8; ++c2) o[c2][i] *= al;
      // Stage P (C-layout) into per-wave LDS for A-fragment re-layout
      Ps[wave][(half * 8 + i) * LDP_S + n] = (bf16)p0;
      Ps[wave][(half * 8 + i) * LDP_S + 16 + n] = (bf16)p1;
    }

    // O += P * V  (P: 16x32 A-frag; V: B-frags from transposed Vst)
    const v16bf pf = load_a_frag(&Ps[wave][0], LDP_S);
    o[0] = wmma_bf16<false, false>(pf, load_bT_frag(Vst, LDB_S), o[0]);
#pragma unroll
    for (int c2 = 1; c2 < 8; ++c2)  // reuse A (pf) on every subsequent op
      o[c2] = wmma_bf16<true, false>(pf, load_bT_frag(Vst + (c2 * 16) * LDB_S, LDB_S), o[c2]);
    __syncthreads();
  }

  // Epilogue: O /= l, store to [B,S,E] bf16
  const int bb = bh >> 4, h = bh & 15;
#pragma unroll
  for (int c2 = 0; c2 < 8; ++c2)
#pragma unroll
    for (int i = 0; i < 8; ++i) {
      const int srow = qr0 + half * 8 + i;
      const int d = c2 * 16 + n;
      const float val = o[c2][i] / lrun[i];
      Om[((size_t)(bb * S_ + srow)) * E_ + h * HD_ + d] = (bf16)val;
    }
}

// ---------------------------------------------------------------------------
// Output projection: [4096,2048] x [2048,2048] + b_out -> f32 d_out
// ---------------------------------------------------------------------------
__global__ __launch_bounds__(128) void gemm_out_kernel(
    const bf16* __restrict__ A, const bf16* __restrict__ Bm,
    const float* __restrict__ bias, float* __restrict__ out) {
  __shared__ bf16 As[64 * LDA_S];
  __shared__ bf16 Bst[64 * LDB_S];
  const v8f zero = {0.f, 0.f, 0.f, 0.f, 0.f, 0.f, 0.f, 0.f};
  v8f acc[2][2] = {{zero, zero}, {zero, zero}};

  const int m0 = blockIdx.y * 64;
  const int n0 = blockIdx.x * 64;
  gemm_tiles(A, E_, Bm, E_, E_, m0, n0, As, Bst, acc);

  const int lane = threadIdx.x & 31;
  const int wave = threadIdx.x >> 5;
  const int wrow = wave >> 1, wcol = wave & 1;
  const int n = lane & 15, half = lane >> 4;
#pragma unroll
  for (int mi = 0; mi < 2; ++mi)
#pragma unroll
    for (int ni = 0; ni < 2; ++ni)
#pragma unroll
      for (int i = 0; i < 8; ++i) {
        const int row = m0 + wrow * 32 + mi * 16 + half * 8 + i;
        const int col = n0 + wcol * 32 + ni * 16 + n;
        out[(size_t)row * E_ + col] = acc[mi][ni][i] + bias[col];
      }
}

// ---------------------------------------------------------------------------
// Launch
// ---------------------------------------------------------------------------
extern "C" void kernel_launch(void* const* d_in, const int* in_sizes, int n_in,
                              void* d_out, int out_size, void* d_ws, size_t ws_size,
                              hipStream_t stream) {
  const float* inX   = (const float*)d_in[0];
  const float* W_qkv = (const float*)d_in[1];
  const float* b_qkv = (const float*)d_in[2];
  const float* W_out = (const float*)d_in[3];
  const float* b_out = (const float*)d_in[4];
  float* out = (float*)d_out;

  constexpr size_t SZ_X    = (size_t)B_ * S_ * E_;        // 8.39M
  constexpr size_t SZ_WQKV = (size_t)E_ * 3 * E_;         // 12.58M
  constexpr size_t SZ_WOUT = (size_t)E_ * E_;             // 4.19M
  constexpr size_t SZ_HEAD = (size_t)B_ * H_ * S_ * HD_;  // 8.39M

  bf16* Xb  = (bf16*)d_ws;
  bf16* Wqb = Xb + SZ_X;
  bf16* Wob = Wqb + SZ_WQKV;
  bf16* Qb  = Wob + SZ_WOUT;
  bf16* Kb  = Qb + SZ_HEAD;
  bf16* Vb  = Kb + SZ_HEAD;
  bf16* Ob  = Vb + SZ_HEAD;

  cvt_f32_bf16<<<2048, 256, 0, stream>>>(inX, Xb, (int)SZ_X);
  cvt_f32_bf16<<<2048, 256, 0, stream>>>(W_qkv, Wqb, (int)SZ_WQKV);
  cvt_f32_bf16<<<2048, 256, 0, stream>>>(W_out, Wob, (int)SZ_WOUT);

  // QKV GEMM: N tiles = 6144/64 = 96, M tiles = 4096/64 = 64
  gemm_qkv_kernel<<<dim3(96, 64), 128, 0, stream>>>(Xb, Wqb, b_qkv, Qb, Kb, Vb);

  // Attention: (S/64, B*H) blocks
  attn_kernel<<<dim3(S_ / 64, B_ * H_), 128, 0, stream>>>(Qb, Kb, Vb, Ob);

  // Output projection: N tiles = 2048/64 = 32, M tiles = 64
  gemm_out_kernel<<<dim3(32, 64), 128, 0, stream>>>(Ob, Wob, b_out, out);
}
